// AIMQuantizerForVJEPA_83227876262437
// MI455X (gfx1250) — matrix operand order repaired and checked by reading, not compiled, not run on hardware
//
#include <hip/hip_runtime.h>

// ---------------------------------------------------------------------------
// AIM residual VQ for V-JEPA on gfx1250 (MI455X)
//   B=16, N=2048, D=1408, P=256, K=(64,128,256), COMMIT=0.25
// GEMMs: bf16x3 split (hi/lo) via V_WMMA_F32_16X16X32_BF16 (fp32-class acc)
// VQ distances: exact fp32 via V_WMMA_F32_16X16X4_F32
// LDS staging of pure-copy tiles via GLOBAL_LOAD_ASYNC_TO_LDS_B128 (ASYNCcnt)
// ---------------------------------------------------------------------------

typedef __attribute__((ext_vector_type(2)))  float        v2f;
typedef __attribute__((ext_vector_type(8)))  float        v8f;
typedef __attribute__((ext_vector_type(16))) __bf16       v16bf;
typedef __attribute__((ext_vector_type(8)))  unsigned int v8u;
typedef __attribute__((ext_vector_type(4)))  int          v4i;

#define BN_ROWS   32768            // B*N
#define DP        1408
#define PP        256
#define K0        64
#define K1        128
#define K2        256
#define KTOT      (K0 + K1 + K2)   // 448

// ---- CDNA5 async LDS copy support (guarded; falls back to ds stores) ------
#if defined(__has_builtin)
#  if __has_builtin(__builtin_amdgcn_global_load_async_to_lds_b128)
#    define HAVE_ASYNC_LDS 1
#  endif
#  if __has_builtin(__builtin_amdgcn_s_wait_asynccnt)
#    define HAVE_WAIT_ASYNC 1
#  endif
#endif
#ifndef HAVE_ASYNC_LDS
#  define HAVE_ASYNC_LDS 0
#endif
#ifndef HAVE_WAIT_ASYNC
#  define HAVE_WAIT_ASYNC 0
#endif

#if HAVE_ASYNC_LDS
typedef __attribute__((address_space(1))) v4i* as1_v4i_ptr;
typedef __attribute__((address_space(3))) v4i* as3_v4i_ptr;

__device__ __forceinline__ void async_cp_b128(const float* g, float* l) {
    // signature: (v4i AS1* gaddr, v4i AS3* lds, imm offset, imm cpol)
    __builtin_amdgcn_global_load_async_to_lds_b128(
        (as1_v4i_ptr)g, (as3_v4i_ptr)l, 0, 0);
}
__device__ __forceinline__ void wait_async0() {
#if HAVE_WAIT_ASYNC
    __builtin_amdgcn_s_wait_asynccnt(0);
#else
    asm volatile("s_wait_asynccnt 0x0" ::: "memory");
#endif
}
#endif

__device__ __forceinline__ unsigned int bf16_rne(float f) {
    unsigned int x = __float_as_uint(f);
    return (x + 0x7FFFu + ((x >> 16) & 1u)) >> 16;   // round-to-nearest-even
}

// ---------------------------------------------------------------------------
// Kernel 0: precompute ||e||^2 for all 448 codewords, zero the loss accum.
// ---------------------------------------------------------------------------
__global__ void prep_kernel(const float* __restrict__ e0,
                            const float* __restrict__ e1,
                            const float* __restrict__ e2,
                            float* __restrict__ en2,
                            float* __restrict__ loss_accum) {
    int t = blockIdx.x * blockDim.x + threadIdx.x;
    if (t == 0) *loss_accum = 0.0f;
    if (t < KTOT) {
        const float* p;
        if (t < K0)            p = e0 + (size_t)t * PP;
        else if (t < K0 + K1)  p = e1 + (size_t)(t - K0) * PP;
        else                   p = e2 + (size_t)(t - K0 - K1) * PP;
        float s = 0.0f;
        #pragma unroll 8
        for (int i = 0; i < PP; ++i) { float v = p[i]; s += v * v; }
        en2[t] = s;
    }
}

// ---------------------------------------------------------------------------
// bf16x3 tiled GEMM with bias:  C[M][N] = A[M][K] * Bm[K][N] + bias[N]
// 256 threads = 8 waves. Tile: BM=128 (16 rows/wave), BN=64, BK=64.
// hi = bf16(a), lo = bf16(a - hi); D += loA*hiB + hiA*loB + hiA*hiB (fp32 acc)
// -> ~1e-5 relative accuracy at 2.67x the K-rate of 16x16x4 f32 WMMA.
// Staging converts fp32->bf16 in registers, so tiles use VGPR path with
// global_prefetch for the next K-tile to hide HBM latency.
// ---------------------------------------------------------------------------
#define GBM 128
#define GBN 64
#define GBK 64
#define GPB 72   // LDS pitch in bf16 elems (144 B rows, 16B-aligned frags)

__global__ __launch_bounds__(256)
void gemm_bias_kernel(const float* __restrict__ A,
                      const float* __restrict__ Bm,
                      const float* __restrict__ bias,
                      float* __restrict__ C,
                      int M, int N, int K) {
    __shared__ unsigned short Ah[GBM * GPB];
    __shared__ unsigned short Al[GBM * GPB];
    __shared__ unsigned short Bh[GBN * GPB];
    __shared__ unsigned short Bl[GBN * GPB];

    const int t    = threadIdx.x;
    const int wave = t >> 5;
    const int lane = t & 31;
    const int lm   = lane & 15;            // M (A frag) / N (B frag)
    const int kh8  = (lane >> 4) << 3;     // A frag K sub-offset: 0 or 8
    const int kh16 = (lane >> 4) << 4;     // (unused)
    const int m0   = blockIdx.x * GBM;
    const int n0   = blockIdx.y * GBN;

    v8f acc[4] = {};

    for (int kb = 0; kb < K; kb += GBK) {
        // prefetch next K-tile (lowered to global_prefetch_b8)
        if (kb + GBK < K) {
            __builtin_prefetch(A + (size_t)(m0 + (t >> 4)) * K + kb + GBK +
                               (t & 15) * 4, 0, 3);
            __builtin_prefetch(Bm + (size_t)(kb + GBK + (t >> 4)) * N + n0 +
                               (t & 15) * 4, 0, 3);
        }
        // --- stage A tile: 128x64 f32 -> bf16 hi/lo. 8 float4/thread ---
        #pragma unroll
        for (int i = 0; i < 8; ++i) {
            int lin = t + i * 256;             // float4 idx, 16 per row
            int r   = lin >> 4;
            int c   = (lin & 15) * 4;
            float4 v = *(const float4*)(A + (size_t)(m0 + r) * K + kb + c);
            unsigned int h0 = bf16_rne(v.x), h1 = bf16_rne(v.y);
            unsigned int h2 = bf16_rne(v.z), h3 = bf16_rne(v.w);
            unsigned int l0 = bf16_rne(v.x - __uint_as_float(h0 << 16));
            unsigned int l1 = bf16_rne(v.y - __uint_as_float(h1 << 16));
            unsigned int l2 = bf16_rne(v.z - __uint_as_float(h2 << 16));
            unsigned int l3 = bf16_rne(v.w - __uint_as_float(h3 << 16));
            uint2 ph = {h0 | (h1 << 16), h2 | (h3 << 16)};
            uint2 pl = {l0 | (l1 << 16), l2 | (l3 << 16)};
            *(uint2*)(&Ah[r * GPB + c]) = ph;
            *(uint2*)(&Al[r * GPB + c]) = pl;
        }
        // --- stage B tile transposed: BT[n][k] = Bm[kb+k][n0+n], hi/lo ---
        #pragma unroll
        for (int i = 0; i < 4; ++i) {
            int lin = t + i * 256;             // float4 idx, 16 per k-row
            int kr  = lin >> 4;
            int nc  = (lin & 15) * 4;
            float4 v = *(const float4*)(Bm + (size_t)(kb + kr) * N + n0 + nc);
            float f[4] = {v.x, v.y, v.z, v.w};
            #pragma unroll
            for (int j = 0; j < 4; ++j) {
                unsigned int h = bf16_rne(f[j]);
                unsigned int l = bf16_rne(f[j] - __uint_as_float(h << 16));
                Bh[(nc + j) * GPB + kr] = (unsigned short)h;
                Bl[(nc + j) * GPB + kr] = (unsigned short)l;
            }
        }
        __syncthreads();

        const int arow = (wave * 16 + lm) * GPB;
        #pragma unroll
        for (int kk = 0; kk < GBK; kk += 32) {
            // A fragments (16-bit 16x32 layout): two 16B reads each
            uint4 ah0 = *(const uint4*)(&Ah[arow + kk + kh8]);
            uint4 ah1 = *(const uint4*)(&Ah[arow + kk + 16 + kh8]);
            uint4 al0 = *(const uint4*)(&Al[arow + kk + kh8]);
            uint4 al1 = *(const uint4*)(&Al[arow + kk + 16 + kh8]);
            v8u aHu = {ah0.x, ah0.y, ah0.z, ah0.w, ah1.x, ah1.y, ah1.z, ah1.w};
            v8u aLu = {al0.x, al0.y, al0.z, al0.w, al1.x, al1.y, al1.z, al1.w};
            v16bf aH = __builtin_bit_cast(v16bf, aHu);
            v16bf aL = __builtin_bit_cast(v16bf, aLu);
            #pragma unroll
            for (int s = 0; s < 4; ++s) {
                const int brow = (s * 16 + lm) * GPB + kk + kh8;
                uint4 bh0 = *(const uint4*)(&Bh[brow]);
                uint4 bh1 = *(const uint4*)(&Bh[brow + 16]);
                uint4 bl0 = *(const uint4*)(&Bl[brow]);
                uint4 bl1 = *(const uint4*)(&Bl[brow + 16]);
                v8u bHu = {bh0.x, bh0.y, bh0.z, bh0.w, bh1.x, bh1.y, bh1.z, bh1.w};
                v8u bLu = {bl0.x, bl0.y, bl0.z, bl0.w, bl1.x, bl1.y, bl1.z, bl1.w};
                v16bf bH = __builtin_bit_cast(v16bf, bHu);
                v16bf bL = __builtin_bit_cast(v16bf, bLu);
                // small terms first, then the dominant product
                acc[s] = __builtin_amdgcn_wmma_f32_16x16x32_bf16(
                    false, aL, false, bH, (short)0, acc[s], false, false);
                acc[s] = __builtin_amdgcn_wmma_f32_16x16x32_bf16(
                    false, aH, false, bL, (short)0, acc[s], false, false);
                acc[s] = __builtin_amdgcn_wmma_f32_16x16x32_bf16(
                    false, aH, false, bH, (short)0, acc[s], false, false);
            }
        }
        __syncthreads();
    }

    // epilogue: C/D layout: VGPR j -> M=j (lanes 0-15), M=j+8 (lanes 16-31)
    const int mb = m0 + wave * 16 + ((lane >> 4) << 3);
    #pragma unroll
    for (int s = 0; s < 4; ++s) {
        int n = n0 + s * 16 + lm;
        float bv = bias[n];
        #pragma unroll
        for (int j = 0; j < 8; ++j) {
            C[(size_t)(mb + j) * N + n] = acc[s][j] + bv;
        }
    }
}

// ---------------------------------------------------------------------------
// Kernel 2: residual VQ, 3 levels. 128 threads = 4 waves, 64 rows per block
// (16 rows per wave). Codebooks staged in LDS in chunks of 64 codewords via
// async-to-LDS (pure fp32 copy, ASYNCcnt-tracked, no VGPR round-trip);
// Gram tiles G = R * E^T via exact fp32 WMMA; argmin on (||e||^2 - 2G);
// residual and combined updated in LDS; loss accumulated from update diff.
// ---------------------------------------------------------------------------
#define VROWS 64
#define RP 260   // pitch in floats for 256-wide rows (16B-aligned, bank-spread)

__global__ __launch_bounds__(128)
void vq_kernel(const float* __restrict__ zproj,
               float* __restrict__ cmb_out,
               const float* __restrict__ e0,
               const float* __restrict__ e1,
               const float* __restrict__ e2,
               const float* __restrict__ en2,
               int* __restrict__ idx_out,
               float* __restrict__ loss_accum) {
    extern __shared__ float smem[];
    float* R   = smem;                    // [64][RP] residual rows
    float* E   = R + VROWS * RP;          // [64][RP] codebook chunk
    float* CMB = E + 64 * RP;             // [64][256] combined accumulation
    float* EN2 = CMB + VROWS * 256;       // [64] ||e||^2 for chunk
    int*  IDXb = (int*)(EN2 + 64);        // [64] selected index per row

    const int t    = threadIdx.x;
    const int wave = t >> 5;
    const int lane = t & 31;
    const int lm   = lane & 15;
    const int kh   = (lane >> 4) << 1;
    const int row0 = blockIdx.x * VROWS;

    // ---- load z_proj rows into R (async copy), zero CMB ----
    #pragma unroll
    for (int i = 0; i < 32; ++i) {
        int lin = t + i * 128;            // float4 index, 64 per row
        int r   = lin >> 6;
        int c   = (lin & 63) * 4;
#if HAVE_ASYNC_LDS
        async_cp_b128(zproj + (size_t)(row0 + r) * PP + c, &R[r * RP + c]);
#else
        float4 v = *(const float4*)(zproj + (size_t)(row0 + r) * PP + c);
        *(float4*)(&R[r * RP + c]) = v;
#endif
        float4 z4 = {0.f, 0.f, 0.f, 0.f};
        *(float4*)(&CMB[r * 256 + c]) = z4;
    }
#if HAVE_ASYNC_LDS
    wait_async0();
#endif
    __syncthreads();

    float lossacc = 0.0f;

    const float* embs[3] = {e0, e1, e2};
    const int    kcbs[3] = {K0, K1, K2};
    const int    eoff[3] = {0, K0, K0 + K1};

    for (int level = 0; level < 3; ++level) {
        const float* EM  = embs[level];
        const int    Kcb = kcbs[level];

        float bestd[8];
        int   bestn[8];
        #pragma unroll
        for (int j = 0; j < 8; ++j) { bestd[j] = 3.4e38f; bestn[j] = 0; }

        for (int chunk = 0; chunk < Kcb; chunk += 64) {
            // stage 64 codewords into LDS (async fp32 copy with pitch pad)
            #pragma unroll
            for (int i = 0; i < 32; ++i) {
                int lin = t + i * 128;
                int r   = lin >> 6;
                int c   = (lin & 63) * 4;
#if HAVE_ASYNC_LDS
                async_cp_b128(EM + (size_t)(chunk + r) * PP + c, &E[r * RP + c]);
#else
                float4 v = *(const float4*)(EM + (size_t)(chunk + r) * PP + c);
                *(float4*)(&E[r * RP + c]) = v;
#endif
            }
            if (t < 64) EN2[t] = en2[eoff[level] + chunk + t];
#if HAVE_ASYNC_LDS
            wait_async0();
#endif
            __syncthreads();

            // Gram tile: G[16 rows][64 cw] for this wave via fp32 WMMA
            v8f acc[4] = {};
            const float* Ar = &R[(wave * 16 + lm) * RP + kh];
            #pragma unroll 4
            for (int kk = 0; kk < PP; kk += 4) {
                v2f a = *(const v2f*)(Ar + kk);
                #pragma unroll
                for (int s = 0; s < 4; ++s) {
                    v2f b = *(const v2f*)(&E[(s * 16 + lm) * RP + kk + kh]);
                    acc[s] = __builtin_amdgcn_wmma_f32_16x16x4_f32(
                        false, a, false, b, (short)0, acc[s], false, false);
                }
            }
            // running argmin on (||e||^2 - 2 G); ||r||^2 is constant over n
            #pragma unroll
            for (int s = 0; s < 4; ++s) {
                float e2 = EN2[s * 16 + lm];
                int   ng = chunk + s * 16 + lm;
                #pragma unroll
                for (int j = 0; j < 8; ++j) {
                    float d = e2 - 2.0f * acc[s][j];
                    if (d < bestd[j]) { bestd[j] = d; bestn[j] = ng; }
                }
            }
            __syncthreads();   // before next chunk overwrites E
        }

        // cross-lane argmin over the 16 lanes of each half (C-layout halves)
        #pragma unroll
        for (int j = 0; j < 8; ++j) {
            float bd = bestd[j];
            int   bi = bestn[j];
            #pragma unroll
            for (int off = 8; off >= 1; off >>= 1) {
                float od = __shfl_xor(bd, off, 32);
                int   oi = __shfl_xor(bi, off, 32);
                if (od < bd || (od == bd && oi < bi)) { bd = od; bi = oi; }
            }
            if (lm == 0) IDXb[wave * 16 + j + ((lane >> 4) << 3)] = bi;
        }
        __syncthreads();

        // write indices for this level
        if (t < VROWS) idx_out[(size_t)level * BN_ROWS + row0 + t] = IDXb[t];

        // residual update + combined accumulation + exact loss term
        for (int m = 0; m < 16; ++m) {
            int row = wave * 16 + m;
            int ci  = IDXb[row];
            const float* er = EM + (size_t)ci * PP;   // L2-resident gather
            #pragma unroll
            for (int tc = 0; tc < 8; ++tc) {
                int col = lane + tc * 32;
                float ev = er[col];
                float rv = R[row * RP + col];
                float df = ev - rv;
                lossacc += df * df;                  // (z_q - residual)^2
                R[row * RP + col]  = rv - ev;        // next residual
                CMB[row * 256 + col] += ev;          // e0+e1+e2
            }
        }
        __syncthreads();
    }

    // ---- write combined rows ----
    #pragma unroll
    for (int i = 0; i < 32; ++i) {
        int lin = t + i * 128;
        int r   = lin >> 6;
        int c   = (lin & 63) * 4;
        *(float4*)(cmb_out + (size_t)(row0 + r) * PP + c) =
            *(float4*)(&CMB[r * 256 + c]);
    }

    // ---- loss: wave reduce then one atomic per wave ----
    float ls = lossacc;
    #pragma unroll
    for (int off = 16; off >= 1; off >>= 1) ls += __shfl_xor(ls, off, 32);
    if (lane == 0) atomicAdd(loss_accum, ls);
}

// ---------------------------------------------------------------------------
// Kernel 4: finalize scalar loss
// ---------------------------------------------------------------------------
__global__ void finalize_kernel(const float* __restrict__ loss_accum,
                                float* __restrict__ out) {
    // vq_loss = COMMIT * sum / (BN*P) / 3
    *out = 0.25f * (*loss_accum) / ((float)BN_ROWS * (float)PP * 3.0f);
}

// ---------------------------------------------------------------------------
extern "C" void kernel_launch(void* const* d_in, const int* in_sizes, int n_in,
                              void* d_out, int out_size, void* d_ws, size_t ws_size,
                              hipStream_t stream) {
    const float* z     = (const float*)d_in[0];
    const float* W_in  = (const float*)d_in[1];
    const float* b_in  = (const float*)d_in[2];
    const float* W_out = (const float*)d_in[3];
    const float* b_out = (const float*)d_in[4];
    const float* e0    = (const float*)d_in[5];
    const float* e1    = (const float*)d_in[6];
    const float* e2    = (const float*)d_in[7];

    // workspace: z_proj/combined (in-place), ||e||^2, loss accumulator
    float* zproj = (float*)d_ws;                       // 32768*256 f32
    float* en2   = zproj + (size_t)BN_ROWS * PP;       // 448 f32
    float* lossa = en2 + KTOT;                         // 1 f32

    // output layout (tuple concatenated flat, all 4-byte elems):
    //   z_q_out [16,2048,1408] f32 | idx0 | idx1 | idx2 (int32) | vq_loss f32
    const size_t zq_elems = (size_t)BN_ROWS * DP;      // 46,137,344
    float* zq_out   = (float*)d_out;
    int*   idx_out  = (int*)d_out + zq_elems;
    float* loss_out = (float*)d_out + zq_elems + (size_t)3 * BN_ROWS;

    // dynamic LDS for VQ kernel: 2*(64*260) + 64*256 + 64 floats + 64 ints
    const int vq_lds = (2 * VROWS * RP + VROWS * 256 + 64) * 4 + 64 * 4;
    (void)hipFuncSetAttribute((const void*)vq_kernel,
                              hipFuncAttributeMaxDynamicSharedMemorySize, vq_lds);

    // 0) ||e||^2 + zero loss accumulator
    prep_kernel<<<2, 256, 0, stream>>>(e0, e1, e2, en2, lossa);

    // 1) z_proj = z @ W_in + b_in    [32768,1408]x[1408,256]  (bf16x3 WMMA)
    gemm_bias_kernel<<<dim3(BN_ROWS / GBM, PP / GBN), 256, 0, stream>>>(
        z, W_in, b_in, zproj, BN_ROWS, PP, DP);

    // 2) residual VQ (3 levels), writes combined in place over zproj
    vq_kernel<<<BN_ROWS / VROWS, 128, vq_lds, stream>>>(
        zproj, zproj, e0, e1, e2, en2, idx_out, lossa);

    // 3) z_q_out = combined @ W_out + b_out   [32768,256]x[256,1408]
    gemm_bias_kernel<<<dim3(BN_ROWS / GBM, DP / GBN), 256, 0, stream>>>(
        zproj, W_out, b_out, zq_out, BN_ROWS, DP, PP);

    // 4) scalar loss
    finalize_kernel<<<1, 1, 0, stream>>>(lossa, loss_out);
}
